// IRNN_54589034332371
// MI455X (gfx1250) — matrix operand order repaired and checked by reading, not compiled, not run on hardware
//
#include <hip/hip_runtime.h>
#include <cstdint>

typedef __bf16 bf16;
typedef __attribute__((ext_vector_type(16))) __bf16 v16bf;
typedef __attribute__((ext_vector_type(8)))  __bf16 v8bf;
typedef __attribute__((ext_vector_type(8)))  float  v8f;

#define T_SEQ 512
#define BATCH 256
#define EMB   512
#define HID   1024
#define HEAD  100
#define NCLS  10

// ---------------------------------------------------------------------------
// Fragment load helpers (CDNA5 wave32 WMMA bf16 layouts, ISA 7.12.2)
// A (16x32, MxK): lane l -> m=l&15, g=l>>4; halves 0-7 = K[g*8 .. g*8+7],
//                 halves 8-15 = K[16+g*8 .. 16+g*8+7]  -> two b128 loads.
// B fragments are pre-swizzled so lane l owns 16 contiguous halves.
// ---------------------------------------------------------------------------
__device__ __forceinline__ v16bf load_a_frag(const bf16* base, int lane,
                                             int kbase, int rowStride) {
  int m = lane & 15, g = lane >> 4;
  const bf16* row = base + m * rowStride + kbase + g * 8;
  v8bf lo = *(const v8bf*)(row);
  v8bf hi = *(const v8bf*)(row + 16);
  v16bf a;
#pragma unroll
  for (int i = 0; i < 8; ++i) { a[i] = lo[i]; a[i + 8] = hi[i]; }
  return a;
}

__device__ __forceinline__ v16bf load_b_frag(const bf16* frag, int lane) {
  const v8bf* p = (const v8bf*)(frag + lane * 16);
  v8bf lo = p[0];
  v8bf hi = p[1];
  v16bf b;
#pragma unroll
  for (int i = 0; i < 8; ++i) { b[i] = lo[i]; b[i + 8] = hi[i]; }
  return b;
}

// ---------------------------------------------------------------------------
// Pack a torch-layout weight [N_out][K_in] (f32) into WMMA B-operand
// fragments (bf16). Fragment f = NT*(K/32) + kb is 32 lanes x 16 halves:
//   lane l, half j  ->  src[(NT*16 + (l&15)) * K + kb*32 + (l>>4)*8 + (j<8 ? j : j+8)]
// One thread per (fragment, lane).
// ---------------------------------------------------------------------------
__global__ void pack_b_frags(const float* __restrict__ src,
                             bf16* __restrict__ dst, int K) {
  int gid  = blockIdx.x * blockDim.x + threadIdx.x;
  int lane = gid & 31;
  int f    = gid >> 5;
  int KB   = K >> 5;
  int kb   = f % KB;
  int NT   = f / KB;
  int n     = NT * 16 + (lane & 15);
  int kbase = kb * 32 + (lane >> 4) * 8;
  const float* s = src + (size_t)n * K + kbase;
  bf16* d = dst + (size_t)f * 512 + lane * 16;
#pragma unroll
  for (int j = 0; j < 16; ++j) d[j] = (bf16)s[j < 8 ? j : j + 8];
}

// ---------------------------------------------------------------------------
// Phase 1: ux[t,b,:] = emb[x_ids[b,t]] @ U^T   (bf16 WMMA, f32 accumulate)
// One block per (t, 16-batch tile): stages the 16 gathered embedding rows in
// LDS (bf16). Loop nest is K-outer / N-inner with an acc[8] register tile so
// each A-fragment is loaded from LDS once per kb and consumed immediately
// (prevents the invariant-hoist + scratch-spill seen in the previous round)
// and is reused by 8 WMMAs. Output is written directly in C-fragment layout.
// ---------------------------------------------------------------------------
__global__ void embed_uproj(const int* __restrict__ x_ids,
                            const float* __restrict__ emb,
                            const bf16* __restrict__ ufrag,
                            bf16* __restrict__ uxfrag) {
  __shared__ __align__(16) bf16 albuf[16 * EMB];   // 16 rows x 512 K, 16 KB
  int tid   = threadIdx.x;                          // 0..255
  int mtile = blockIdx.x;                           // 0..8191
  int t  = mtile >> 4;
  int bt = mtile & 15;

  // Stage A: gather embedding rows, convert to bf16 (coalesced over columns)
#pragma unroll
  for (int i = 0; i < 32; ++i) {
    int idx = tid + i * 256;           // 0..8191 over 16x512
    int row = idx >> 9;
    int col = idx & 511;
    int b   = bt * 16 + row;
    int id  = x_ids[b * T_SEQ + t];
    albuf[row * EMB + col] = (bf16)emb[(size_t)id * EMB + col];
  }
  __syncthreads();

  int lane = tid & 31;
  int wave = tid >> 5;                               // 0..7

  v8f acc[8];                                        // wave's 8 N-tiles
#pragma unroll
  for (int u = 0; u < 8; ++u)
#pragma unroll
    for (int i = 0; i < 8; ++i) acc[u][i] = 0.0f;

#pragma unroll 1
  for (int kb = 0; kb < EMB / 32; ++kb) {            // 16 k-iters, rolled
    v16bf a = load_a_frag(albuf, lane, kb * 32, EMB);
#pragma unroll
    for (int u = 0; u < 8; ++u) {
      int NT = u * 8 + wave;                         // 0..63
      v16bf b = load_b_frag(ufrag + ((size_t)NT * (EMB / 32) + kb) * 512, lane);
      acc[u] = __builtin_amdgcn_wmma_f32_16x16x32_bf16(false, a, false, b,
                                                       (short)0, acc[u],
                                                       false, false);
    }
  }

  // Store in C-fragment layout: frag = (t*16+bt)*64 + NT, per-lane 8 halves
#pragma unroll
  for (int u = 0; u < 8; ++u) {
    int NT = u * 8 + wave;
    bf16* d = uxfrag + (((size_t)(t * 16 + bt)) * 64 + NT) * 256 + lane * 8;
    v8bf o;
#pragma unroll
    for (int r = 0; r < 8; ++r) o[r] = (bf16)acc[u][r];
    *(v8bf*)d = o;
  }
}

// ---------------------------------------------------------------------------
// Phase 2+3: Elman recurrence + heads. One block per 16-batch tile (h rows
// are batch-independent, so no cross-workgroup sync is ever needed).
// 16 waves; wave w owns N-columns [w*64, w*64+64) = 4 WMMA tiles.
// h double-buffered in LDS (bf16, 2 x 32 KB); W streamed from L2 each step
// in fragment-native layout; ux fragments stream from HBM once.
// ---------------------------------------------------------------------------
__global__ void irnn_recurrence(const bf16* __restrict__ wfrag,
                                const bf16* __restrict__ uxfrag,
                                const float* __restrict__ W_b,
                                const float* __restrict__ h1_w,
                                const float* __restrict__ h1_b,
                                const float* __restrict__ h2_w,
                                const float* __restrict__ h2_b,
                                float* __restrict__ out) {
  __shared__ __align__(16) bf16 hbuf[2][16 * HID];  // 64 KB double buffer
  __shared__ float zbuf[16 * HEAD];                 // 6.4 KB

  int tid  = threadIdx.x;       // 0..511
  int lane = tid & 31;
  int wave = tid >> 5;          // 0..15
  int bt   = blockIdx.x;        // 0..15
  int g    = lane >> 4;

  // h0 = 0
#pragma unroll
  for (int i = 0; i < 32; ++i) hbuf[0][tid + i * 512] = (bf16)0.0f;
  __syncthreads();

  // Recurrent bias, one scalar per owned N-tile (C-layout: n = NT*16+(l&15))
  float wb[4];
#pragma unroll
  for (int nt = 0; nt < 4; ++nt)
    wb[nt] = W_b[(wave * 4 + nt) * 16 + (lane & 15)];

  int cur = 0;
  for (int t = 0; t < T_SEQ; ++t) {
    v8f acc[4];
#pragma unroll
    for (int nt = 0; nt < 4; ++nt)
#pragma unroll
      for (int i = 0; i < 8; ++i) acc[nt][i] = 0.0f;

    const bf16* hb = hbuf[cur];
#pragma unroll 2
    for (int kb = 0; kb < HID / 32; ++kb) {          // 32 k-iters
      v16bf a = load_a_frag(hb, lane, kb * 32, HID);
#pragma unroll
      for (int nt = 0; nt < 4; ++nt) {
        int NTg = wave * 4 + nt;
        v16bf b = load_b_frag(wfrag + ((size_t)NTg * 32 + kb) * 512, lane);
        acc[nt] = __builtin_amdgcn_wmma_f32_16x16x32_bf16(false, a, false, b,
                                                          (short)0, acc[nt],
                                                          false, false);
      }
    }

    // Epilogue: + ux_t + b, relu, write next h buffer
    int nxt = cur ^ 1;
    bf16* hn = hbuf[nxt];
#pragma unroll
    for (int nt = 0; nt < 4; ++nt) {
      int NTg = wave * 4 + nt;
      const v8bf* up = (const v8bf*)(uxfrag +
          (((size_t)(t * 16 + bt)) * 64 + NTg) * 256 + lane * 8);
      v8bf u = *up;
      int n = NTg * 16 + (lane & 15);
#pragma unroll
      for (int r = 0; r < 8; ++r) {
        float v = acc[nt][r] + (float)u[r] + wb[nt];
        v = v > 0.0f ? v : 0.0f;
        hn[(g * 8 + r) * HID + n] = (bf16)v;         // C-layout scatter to LDS
      }
    }
    __syncthreads();
    cur = nxt;
  }

  // Phase 3a: z = relu(h @ h1^T + b1)   [16 x 100]
  const bf16* hf = hbuf[cur];
  for (int i = tid; i < 16 * HEAD; i += 512) {
    int b  = i / HEAD;
    int hd = i % HEAD;
    const float* w = h1_w + (size_t)hd * HID;
    float s = h1_b[hd];
    for (int k = 0; k < HID; ++k) s += (float)hf[b * HID + k] * w[k];
    zbuf[b * HEAD + hd] = s > 0.0f ? s : 0.0f;
  }
  __syncthreads();

  // Phase 3b: out = z @ h2^T + b2       [16 x 10]
  if (tid < 16 * NCLS) {
    int b = tid / NCLS;
    int c = tid % NCLS;
    const float* w = h2_w + (size_t)c * HEAD;
    float s = h2_b[c];
    for (int k = 0; k < HEAD; ++k) s += zbuf[b * HEAD + k] * w[k];
    out[(bt * 16 + b) * NCLS + c] = s;
  }
}

// ---------------------------------------------------------------------------
extern "C" void kernel_launch(void* const* d_in, const int* in_sizes, int n_in,
                              void* d_out, int out_size, void* d_ws,
                              size_t ws_size, hipStream_t stream) {
  const int*   x_ids = (const int*)d_in[0];
  const float* emb   = (const float*)d_in[1];
  const float* U_w   = (const float*)d_in[2];
  const float* W_w   = (const float*)d_in[3];
  const float* W_b   = (const float*)d_in[4];
  const float* h1_w  = (const float*)d_in[5];
  const float* h1_b  = (const float*)d_in[6];
  const float* h2_w  = (const float*)d_in[7];
  const float* h2_b  = (const float*)d_in[8];
  float* out = (float*)d_out;

  uint8_t* ws = (uint8_t*)d_ws;
  // Workspace map:
  //   [0,              256 MB)  ux fragments (bf16, C-layout)
  //   [256 MB,         +2 MB )  W fragments  (bf16, B-layout)
  //   [256 MB + 2 MB,  +1 MB )  U fragments  (bf16, B-layout)
  bf16* uxfrag = (bf16*)ws;
  bf16* wfrag  = (bf16*)(ws + (size_t)256 * 1024 * 1024);
  bf16* ufrag  = (bf16*)(ws + (size_t)256 * 1024 * 1024 + 2 * 1024 * 1024);

  // Pack W [1024x1024] -> 2048 fragments (65536 threads)
  pack_b_frags<<<256, 256, 0, stream>>>(W_w, wfrag, HID);
  // Pack U [1024x512]  -> 1024 fragments (32768 threads)
  pack_b_frags<<<128, 256, 0, stream>>>(U_w, ufrag, EMB);
  // Phase 1: 8192 (t, batch-tile) blocks x 8 waves
  embed_uproj<<<8192, 256, 0, stream>>>(x_ids, emb, ufrag, uxfrag);
  // Phase 2+3: 16 batch-tile blocks x 16 waves, T=512 steps each
  irnn_recurrence<<<16, 512, 0, stream>>>(wfrag, uxfrag, W_b, h1_w, h1_b,
                                          h2_w, h2_b, out);
}
